// Network_34385508172429
// MI455X (gfx1250) — matrix operand order repaired and checked by reading, not compiled, not run on hardware
//
#include <hip/hip_runtime.h>
#include <hip/hip_bf16.h>

typedef __attribute__((ext_vector_type(16))) __bf16 v16bf;
typedef __attribute__((ext_vector_type(8)))  __bf16 v8bf;
typedef __attribute__((ext_vector_type(8)))  float  v8f;
typedef unsigned int u32x4 __attribute__((ext_vector_type(4)));
typedef int          i32x4 __attribute__((ext_vector_type(4)));
typedef int          i32x8 __attribute__((ext_vector_type(8)));

#define NPIX 128
#define HALO 2
#define WROW 132                  // 128 + 2*HALO
#define CP   32                   // padded channel count (WMMA K slice)
#define SPIX 16384                // 128*128
#define NCOT 11
#define CHID 24
#define CIN  18
#define COUT 12
#define QDIM 8
#define GSTR ((long)WROW*WROW*CP) // elems per halo image
#define TILE_ELEMS (5 * WROW * CP)   // bf16 elems staged per block (5 halo rows)

#if __has_builtin(__builtin_amdgcn_tensor_load_to_lds)
#define HAVE_TDM 1
#else
#define HAVE_TDM 0
#endif

#define WMMA_BF16(A,B,C) \
  __builtin_amdgcn_wmma_f32_16x16x32_bf16(false,(A),false,(B),(short)0,(C),false,false)

__device__ __forceinline__ v16bf load_afrag(const __bf16* p) {
  // p points at this lane's first K element; K halves at +0 and +16 elems
  v8bf lo = *(const v8bf*)(p);
  v8bf hi = *(const v8bf*)(p + 16);
  return __builtin_shufflevector(lo, hi, 0,1,2,3,4,5,6,7,8,9,10,11,12,13,14,15);
}

// ---------------------------------------------------------------------------
// Implicit-GEMM 5x5 SAME conv via v_wmma_f32_16x16x32_bf16, with the 5 input
// halo rows staged once per block into LDS by the Tensor Data Mover.
// src : bf16 halo image [WROW][WROW][CP]  (+ j*srcStride for batch)
// wt  : bf16 packed weights [25][32 rows(M=co)][32 cols(K=ci)]
// MODE 0: out = relu(acc+bias)        -> dstHalo (bf16 halo layout)
// MODE 1: out = acc * (mask>0)        -> dstHalo
// MODE 2: out = acc+bias (co<coLimit) -> dstF32  [co][128][128]
// ---------------------------------------------------------------------------
template <int NT, int MODE>
__launch_bounds__(256)
__global__ void conv5x5_wmma(const __bf16* __restrict__ src,
                             const __bf16* __restrict__ wt,
                             const float*  __restrict__ bias,
                             const __bf16* __restrict__ mask,
                             __bf16* __restrict__ dstHalo,
                             float*  __restrict__ dstF32,
                             int coLimit, long srcStride, long dstStride)
{
  const int y    = blockIdx.x;          // output row
  const int j    = blockIdx.y;          // batch (cotangent) index
  const int lane = threadIdx.x & 31;
  const int wave = threadIdx.x >> 5;
  const int n    = lane & 15;           // N index (pixel column within tile)
  const int x    = wave * 16 + n;       // output column
  const int hiA  = (lane & 16) ? 16 : 0;// A-frag K-half offset
  const int hiB  = (lane & 16) ? 16 : 0;// B-frag K-half offset

  // ---- stage 5 contiguous halo rows (y..y+4) into LDS ----
  __shared__ __align__(64) __bf16 tile[TILE_ELEMS];
  const __bf16* srcRows = src + (long)j * srcStride + (long)y * WROW * CP;
#if HAVE_TDM
  if (wave == 0) {
    const unsigned ldsOff = (unsigned)(uintptr_t)(void*)tile;       // addr[31:0] = LDS offset
    const unsigned long long ga = (unsigned long long)(uintptr_t)srcRows;
    const unsigned nDw = (unsigned)(TILE_ELEMS * 2 / 4);            // 10560 dwords, 1-row tile
    u32x4 g0;
    g0[0] = 1u;                                                     // count=1, user descriptor
    g0[1] = ldsOff;                                                 // D#.lds_addr
    g0[2] = (unsigned)ga;                                           // D#.global_addr[31:0]
    g0[3] = (unsigned)((ga >> 32) & 0x1FFFFFFu) | (2u << 30);       // addr[56:32] | type=2
    i32x8 g1;
    g1[0] = (int)(2u << 16);                                        // wg_mask=0, data_size=4B
    g1[1] = (int)((nDw & 0xFFFFu) << 16);                           // tensor_dim0[15:0]
    g1[2] = (int)((nDw >> 16) | (1u << 16));                        // tensor_dim0[31:16] | tensor_dim1=1
    g1[3] = (int)((nDw & 0xFFFFu) << 16);                           // tile_dim0 = nDw
    g1[4] = 0;                                                      // tile_dim1/2 unused
    g1[5] = (int)nDw;                                               // tensor_dim0_stride lo
    g1[6] = 0;
    g1[7] = 0;
    i32x4 gz = {0, 0, 0, 0};
#if defined(__clang_major__) && (__clang_major__ >= 23)
    i32x8 gz8 = {0, 0, 0, 0, 0, 0, 0, 0};
    __builtin_amdgcn_tensor_load_to_lds(g0, g1, gz, gz, gz8, 0);
#else
    __builtin_amdgcn_tensor_load_to_lds(g0, g1, gz, gz, 0);
#endif
    __builtin_amdgcn_s_wait_tensorcnt(0);
  }
#else
  for (int i = threadIdx.x; i < TILE_ELEMS / 8; i += 256)
    ((v8bf*)tile)[i] = ((const v8bf*)srcRows)[i];
#endif
  __syncthreads();

  v8f acc0 = {}; v8f acc1 = {};
  #pragma unroll
  for (int tap = 0; tap < 25; ++tap) {
    const int dy = tap / 5, dx = tap - dy * 5;
    // B fragment: 32 contiguous bf16 channels of staged pixel (dy, x+dx)
    const __bf16* bp = tile + ((dy * WROW) + (x + dx)) * CP + hiB;
    v16bf B = *(const v16bf*)bp;
    // A fragment(s): row M = co = t*16 + (lane&15), K = ci
    const __bf16* ap = wt + ((long)tap * 32 + n) * CP + hiA;
    v16bf A0 = *(const v16bf*)ap;
    acc0 = WMMA_BF16(A0, B, acc0);
    if (NT > 1) {
      v16bf A1 = *(const v16bf*)(ap + 16 * CP);
      acc1 = WMMA_BF16(A1, B, acc1);
    }
  }

  // Epilogue. C/D layout: VGPR i -> M = i + ((lane&16)?8:0), N = lane&15.
  #pragma unroll
  for (int t = 0; t < NT; ++t) {
    v8f acc;
    if (t == 0) acc = acc0; else acc = acc1;
    const int cobase = t * 16 + ((lane & 16) ? 8 : 0);
    if (MODE == 2) {
      float* dp = dstF32 + (long)j * dstStride;
      #pragma unroll
      for (int i = 0; i < 8; ++i) {
        int co = cobase + i;
        if (co < coLimit)
          dp[(long)co * SPIX + y * NPIX + x] = acc[i] + (bias ? bias[co] : 0.f);
      }
    } else {
      const long pofs = ((long)(y + HALO) * WROW + (x + HALO)) * CP + cobase;
      v8bf o;
      if (MODE == 1) {
        v8bf mv = *(const v8bf*)(mask + pofs);
        #pragma unroll
        for (int i = 0; i < 8; ++i) {
          float f = (cobase + i < coLimit) ? acc[i] : 0.f;
          o[i] = (__bf16)(((float)mv[i] > 0.f) ? f : 0.f);
        }
      } else {
        #pragma unroll
        for (int i = 0; i < 8; ++i) {
          int co = cobase + i;
          float f = acc[i];
          if (bias && co < coLimit) f += bias[co];
          if (co >= coLimit) f = 0.f;
          o[i] = (__bf16)(f > 0.f ? f : 0.f);
        }
      }
      *(v8bf*)(dstHalo + (long)j * dstStride + pofs) = o;
    }
  }
}

// ---------------------------------------------------------------------------
// g4 = m4 * conv5^T(one-hot-channel all-ones cotangent): closed form border sum
// ---------------------------------------------------------------------------
__global__ void g4_init(const float* __restrict__ W5,
                        const __bf16* __restrict__ A4,
                        __bf16* __restrict__ G)
{
  const int x = threadIdx.x;            // 128
  const int y = blockIdx.x;             // 128
  const int j = blockIdx.y;             // 11 cotangents
  const int ch = (j < 3) ? j : j + 1;   // idx = [0,1,2,4..11]
  const long pofs = ((long)(y + HALO) * WROW + (x + HALO)) * CP;
  __bf16* gp = G + (long)j * GSTR + pofs;
  const __bf16* mp = A4 + pofs;
  for (int ci = 0; ci < CP; ++ci) {
    float s = 0.f;
    if (ci < CHID) {
      for (int dy = 0; dy < 5; ++dy) {
        int yy = y + dy - 2; if (yy < 0 || yy >= NPIX) continue;
        for (int dx = 0; dx < 5; ++dx) {
          int xx = x + dx - 2; if (xx < 0 || xx >= NPIX) continue;
          s += W5[(((long)ch * CHID + ci) * 5 + (4 - dy)) * 5 + (4 - dx)];
        }
      }
      if (!((float)mp[ci] > 0.f)) s = 0.f;
    }
    gp[ci] = (__bf16)s;
  }
}

// ---------------------------------------------------------------------------
// Pack f32 weights -> bf16 [25 taps][32 M rows][32 K cols] (optionally
// transposed + spatially flipped for the VJP convs).
// ---------------------------------------------------------------------------
__global__ void prep_weights(const float* __restrict__ W, __bf16* __restrict__ out,
                             int Co, int Ci, int transpose)
{
  int tid = blockIdx.x * blockDim.x + threadIdx.x;
  if (tid >= 25 * 32 * 32) return;
  int tap = tid >> 10, r = (tid >> 5) & 31, k = tid & 31;
  int dy = tap / 5, dx = tap - dy * 5;
  float v = 0.f;
  if (!transpose) { if (r < Co && k < Ci) v = W[(((long)r * Ci + k) * 5 + dy) * 5 + dx]; }
  else            { if (k < Co && r < Ci) v = W[(((long)k * Ci + r) * 5 + (4 - dy)) * 5 + (4 - dx)]; }
  out[tid] = (__bf16)v;
}

// Pack x,y,u into the bf16 halo input image (ch 0..17, pad to 32).
__global__ void pack_input(const float* __restrict__ xx, const float* __restrict__ yy,
                           const float* __restrict__ uu, __bf16* __restrict__ A0)
{
  int tid = blockIdx.x * blockDim.x + threadIdx.x;
  if (tid >= WROW * WROW) return;
  int py = tid / WROW, px = tid - py * WROW;
  int iy = py - HALO, ix = px - HALO;
  bool in = (iy >= 0 && iy < NPIX && ix >= 0 && ix < NPIX);
  int p = iy * NPIX + ix;
  __bf16* dp = A0 + (long)tid * CP;
  for (int c = 0; c < CP; ++c) {
    float v = 0.f;
    if (in) {
      if (c == 0)      v = xx[p];
      else if (c == 1) v = yy[p];
      else if (c < CIN) v = uu[(long)(c - 2) * SPIX + p];
    }
    dp[c] = (__bf16)v;
  }
}

__global__ void zero_u32(unsigned* __restrict__ p, long n)
{
  long i = (long)blockIdx.x * blockDim.x + threadIdx.x;
  if (i < n) p[i] = 0u;
}

// ---------------------------------------------------------------------------
// pde = (kappa_x+v1)*u_rk_x + (kappa_y+v2)*u_rk_y + f   (Hessian terms are 0
// exactly: the net is piecewise-linear in x,y). Then the RK mix + assembly.
// ---------------------------------------------------------------------------
__global__ void pde_rk(const float* __restrict__ out5, const float* __restrict__ gxy,
                       const float* __restrict__ rkA, const float* __restrict__ rkb,
                       float* __restrict__ out)
{
  int p = blockIdx.x * blockDim.x + threadIdx.x;
  if (p >= SPIX) return;
  const float DT = 1.0f;
  float kap = out5[0 * SPIX + p], v1 = out5[1 * SPIX + p];
  float v2  = out5[2 * SPIX + p], f  = out5[3 * SPIX + p];
  float gx0 = gxy[(0 * 2 + 0) * SPIX + p], gy0 = gxy[(0 * 2 + 1) * SPIX + p];
  float gx1 = gxy[(1 * 2 + 0) * SPIX + p], gy1 = gxy[(1 * 2 + 1) * SPIX + p];
  float gx2 = gxy[(2 * 2 + 0) * SPIX + p], gy2 = gxy[(2 * 2 + 1) * SPIX + p];
  float pde[QDIM], urk[QDIM];
  #pragma unroll
  for (int q = 0; q < QDIM; ++q) {
    urk[q] = out5[(4 + q) * SPIX + p];
    float ux = gxy[((3 + q) * 2 + 0) * SPIX + p];
    float uy = gxy[((3 + q) * 2 + 1) * SPIX + p];
    pde[q] = (gx0 + v1) * ux + (gy0 + v2) * uy + f;
  }
  #pragma unroll
  for (int i = 0; i < QDIM; ++i) {
    float s = 0.f, s2 = 0.f;
    #pragma unroll
    for (int q = 0; q < QDIM; ++q) {
      float a = rkA[i * QDIM + q];
      s  += a * pde[q];
      s2 += (a - rkb[q]) * pde[q];
    }
    out[i * SPIX + p]       = urk[i] + DT * s;
    out[(8 + i) * SPIX + p] = urk[i] + DT * s2;
  }
  out[16 * SPIX + p] = kap; out[17 * SPIX + p] = gx0; out[18 * SPIX + p] = gy0;
  out[19 * SPIX + p] = v1;  out[20 * SPIX + p] = gx1; out[21 * SPIX + p] = gy1;
  out[22 * SPIX + p] = v2;  out[23 * SPIX + p] = gx2; out[24 * SPIX + p] = gy2;
  out[25 * SPIX + p] = f;
}

// ---------------------------------------------------------------------------
extern "C" void kernel_launch(void* const* d_in, const int* in_sizes, int n_in,
                              void* d_out, int out_size, void* d_ws, size_t ws_size,
                              hipStream_t stream) {
  const float* X  = (const float*)d_in[0];
  const float* Y  = (const float*)d_in[1];
  const float* U  = (const float*)d_in[2];
  const float* W1 = (const float*)d_in[3];  const float* B1 = (const float*)d_in[4];
  const float* W2 = (const float*)d_in[5];  const float* B2 = (const float*)d_in[6];
  const float* W3 = (const float*)d_in[7];  const float* B3 = (const float*)d_in[8];
  const float* W4 = (const float*)d_in[9];  const float* B4 = (const float*)d_in[10];
  const float* W5 = (const float*)d_in[11]; const float* B5 = (const float*)d_in[12];
  const float* RKA = (const float*)d_in[13];
  const float* RKB = (const float*)d_in[14];
  float* OUT = (float*)d_out;

  // ---- workspace carve-up (bf16 halo region first, then f32) ----
  __bf16* halo = (__bf16*)d_ws;           // A0..A4 + GA(11) + GB(11) = 27*GSTR
  __bf16* A0 = halo + 0 * GSTR;
  __bf16* A1 = halo + 1 * GSTR;
  __bf16* A2 = halo + 2 * GSTR;
  __bf16* A3 = halo + 3 * GSTR;
  __bf16* A4 = halo + 4 * GSTR;
  __bf16* GA = halo + 5 * GSTR;
  __bf16* GB = GA + (long)NCOT * GSTR;
  __bf16* Wb  = halo + 27 * GSTR;         // 9 packed weight blocks of 25600
  __bf16* Wb1 = Wb + 0 * 25600; __bf16* Wb2 = Wb + 1 * 25600;
  __bf16* Wb3 = Wb + 2 * 25600; __bf16* Wb4 = Wb + 3 * 25600;
  __bf16* Wb5 = Wb + 4 * 25600;
  __bf16* Wt4 = Wb + 5 * 25600; __bf16* Wt3 = Wb + 6 * 25600;
  __bf16* Wt2 = Wb + 7 * 25600; __bf16* Wt1 = Wb + 8 * 25600;
  float* OUT5 = (float*)(Wb + 9 * 25600); // 12*SPIX
  float* GXY  = OUT5 + 12L * SPIX;        // 11*2*SPIX

  // ---- zero all halo buffers (borders + pad channels must be 0) ----
  long zw = (27L * GSTR * 2) / 4;
  zero_u32<<<(unsigned)((zw + 255) / 256), 256, 0, stream>>>((unsigned*)halo, zw);

  pack_input<<<(WROW * WROW + 255) / 256, 256, 0, stream>>>(X, Y, U, A0);

  prep_weights<<<100, 256, 0, stream>>>(W1, Wb1, CHID, CIN,  0);
  prep_weights<<<100, 256, 0, stream>>>(W2, Wb2, CHID, CHID, 0);
  prep_weights<<<100, 256, 0, stream>>>(W3, Wb3, CHID, CHID, 0);
  prep_weights<<<100, 256, 0, stream>>>(W4, Wb4, CHID, CHID, 0);
  prep_weights<<<100, 256, 0, stream>>>(W5, Wb5, COUT, CHID, 0);
  prep_weights<<<100, 256, 0, stream>>>(W4, Wt4, CHID, CHID, 1);
  prep_weights<<<100, 256, 0, stream>>>(W3, Wt3, CHID, CHID, 1);
  prep_weights<<<100, 256, 0, stream>>>(W2, Wt2, CHID, CHID, 1);
  prep_weights<<<100, 256, 0, stream>>>(W1, Wt1, CHID, CIN,  1);

  // ---- forward: h1..h4 (relu, bf16 halo), out5 (f32) ----
  dim3 cg(NPIX, 1);
  conv5x5_wmma<2,0><<<cg, 256, 0, stream>>>(A0, Wb1, B1, nullptr, A1, nullptr, CHID, 0, 0);
  conv5x5_wmma<2,0><<<cg, 256, 0, stream>>>(A1, Wb2, B2, nullptr, A2, nullptr, CHID, 0, 0);
  conv5x5_wmma<2,0><<<cg, 256, 0, stream>>>(A2, Wb3, B3, nullptr, A3, nullptr, CHID, 0, 0);
  conv5x5_wmma<2,0><<<cg, 256, 0, stream>>>(A3, Wb4, B4, nullptr, A4, nullptr, CHID, 0, 0);
  conv5x5_wmma<1,2><<<cg, 256, 0, stream>>>(A4, Wb5, B5, nullptr, nullptr, OUT5, COUT, 0, 0);

  // ---- backward for 11 cotangents (batched over gridDim.y) ----
  g4_init<<<dim3(NPIX, NCOT), NPIX, 0, stream>>>(W5, A4, GA);
  dim3 cgb(NPIX, NCOT);
  conv5x5_wmma<2,1><<<cgb, 256, 0, stream>>>(GA, Wt4, nullptr, A3, GB, nullptr, CHID, GSTR, GSTR);
  conv5x5_wmma<2,1><<<cgb, 256, 0, stream>>>(GB, Wt3, nullptr, A2, GA, nullptr, CHID, GSTR, GSTR);
  conv5x5_wmma<2,1><<<cgb, 256, 0, stream>>>(GA, Wt2, nullptr, A1, GB, nullptr, CHID, GSTR, GSTR);
  conv5x5_wmma<1,2><<<cgb, 256, 0, stream>>>(GB, Wt1, nullptr, nullptr, nullptr, GXY, 2, GSTR, 2L * SPIX);

  // ---- PDE + RK mix + output assembly (26 x 128 x 128 f32) ----
  pde_rk<<<(SPIX + 255) / 256, 256, 0, stream>>>(OUT5, GXY, RKA, RKB, OUT);
}
